// SpatialDecoder_4492535792211
// MI455X (gfx1250) — compile-verified, matrix operand order checked by reading, not run on hardware
//
#include <hip/hip_runtime.h>
#include <hip/hip_bf16.h>

typedef __attribute__((ext_vector_type(16))) _Float16 v16h;
typedef __attribute__((ext_vector_type(8)))  _Float16 v8h;
typedef __attribute__((ext_vector_type(8)))  float    v8f;
typedef __attribute__((ext_vector_type(4)))  int      v4i;

#define BROWS   524288
#define DIN     32
#define HID     128
#define LD      16
#define NE      512
#define NTILES  (NE/16)                 // 32 codebook column tiles
#define WAVES   8
#define ROWS_PER_BLOCK (WAVES*16)       // 128 rows per block
#define NBLOCKS (BROWS/ROWS_PER_BLOCK)  // 4096

// ---- d_out layout (tuple concatenated flat, all float32) ----
#define OUT_ZQ   0
#define OUT_SC   ((size_t)BROWS*LD)
#define OUT_RS   (OUT_SC + BROWS)
#define OUT_LOSS (OUT_RS + BROWS)
#define OUT_IDS  (OUT_LOSS + 1)

// ---- workspace layout ----
// f16 pre-swizzled B-matrix tiles (units: halves)
#define WH_DECW1 0          // 8 tiles * 1 kb * 32 lanes * 16 = 4096
#define WH_SCW1  4096
#define WH_RSW1  8192
#define WH_DECW2 12288      // 1 tile * 4 kb * 512 = 2048
#define WH_CB    14336      // 32 tiles * 1 kb * 512 = 16384
#define WH_TOTAL 30720
#define WSB_CSQ   ((size_t)WH_TOTAL*2)  // 512 floats of ||c||^2, byte offset 61440
#define WSB_STAGE (WSB_CSQ + NE*4)      // bytes staged to LDS per block: 63488
#define WSB_PART  WSB_STAGE             // NBLOCKS block-partial loss sums after stage region

// ds_swizzle, group-of-32 XOR mode: offset = (xor<<10) | (or<<5) | and(0x1F)
#define SWZ_I(v, MASK) __builtin_amdgcn_ds_swizzle((v), (((MASK)<<10)|0x1F))
#define SWZ_F(v, MASK) __int_as_float(SWZ_I(__float_as_int(v), (MASK)))

// ---------------------------------------------------------------------------
// Prep: pack f32 row-major (K x N) weight into f16 WMMA B-matrix tiles.
// Tile (n, kb) covers cols [n*16, n*16+16), K rows [kb*32, kb*32+32).
// Lane layout per ISA 7.12.2: lane<16 -> col n*16+lane, K elems {0..7,16..23};
// lane>=16 -> same cols, K elems {8..15,24..31}. K beyond real K is zero (pad).
// ---------------------------------------------------------------------------
__device__ inline void pack_tile(_Float16* dst, const float* W, int K, int N, int idx) {
  int j    = idx & 15;
  int lane = (idx >> 4) & 31;
  int blk  = idx >> 9;                    // n*KBc + kb
  int KBc  = (K + 31) >> 5;
  int n    = blk / KBc;
  int kb   = blk - n * KBc;
  int h    = lane >> 4;
  int c    = n * 16 + (lane & 15);
  int k    = kb * 32 + ((j < 8) ? (h * 8 + j) : (16 + h * 8 + (j - 8)));
  float v  = (k < K) ? W[k * N + c] : 0.0f;
  dst[idx] = (_Float16)v;
}

__global__ void prep_kernel(const float* __restrict__ dec_w1, const float* __restrict__ sc_w1,
                            const float* __restrict__ rs_w1,  const float* __restrict__ dec_w2,
                            const float* __restrict__ codebook,
                            _Float16* __restrict__ wsH, float* __restrict__ csq) {
  const int total = WH_TOTAL + NE;
  for (int i = blockIdx.x * blockDim.x + threadIdx.x; i < total; i += gridDim.x * blockDim.x) {
    if (i < WH_SCW1)        pack_tile(wsH + WH_DECW1, dec_w1,   DIN, HID, i - WH_DECW1);
    else if (i < WH_RSW1)   pack_tile(wsH + WH_SCW1,  sc_w1,    DIN, HID, i - WH_SCW1);
    else if (i < WH_DECW2)  pack_tile(wsH + WH_RSW1,  rs_w1,    DIN, HID, i - WH_RSW1);
    else if (i < WH_CB)     pack_tile(wsH + WH_DECW2, dec_w2,   HID, LD,  i - WH_DECW2);
    else if (i < WH_TOTAL)  pack_tile(wsH + WH_CB,    codebook, LD,  NE,  i - WH_CB);
    else {
      int e = i - WH_TOTAL;
      float s = 0.0f;
      #pragma unroll
      for (int l = 0; l < LD; ++l) { float c = codebook[l * NE + e]; s += c * c; }
      csq[e] = s;
    }
  }
}

// ---------------------------------------------------------------------------
// Main fused kernel: one wave owns 16 rows. 8 waves / 256 threads per block.
// All B-matrix tiles + ||c||^2 staged into LDS once per block via the CDNA5
// async global->LDS path (ASYNCcnt), then consumed with ds_load_b128.
// ---------------------------------------------------------------------------
__global__ __launch_bounds__(256, 1) void vq_kernel(
    const float* __restrict__ z, const float* __restrict__ codebook,
    const float* __restrict__ dec_b1, const float* __restrict__ dec_b2,
    const float* __restrict__ sc_b1,  const float* __restrict__ sc_w2, const float* __restrict__ sc_b2,
    const float* __restrict__ rs_b1,  const float* __restrict__ rs_w2, const float* __restrict__ rs_b2,
    const _Float16* __restrict__ wsH, float* __restrict__ out, float* __restrict__ partials) {

  __shared__ __align__(16) unsigned char stage[WSB_STAGE]; // packed weights + cb + csq (62KB)
  __shared__ __align__(16) _Float16 hbuf[WAVES][16][HID];  // hidden activations, f16
  __shared__ __align__(16) _Float16 zbuf[WAVES][16][LD];   // decoded latents, f16
  __shared__ float wsum[WAVES];

  const int tid  = threadIdx.x;
  const int wave = tid >> 5;
  const int lane = tid & 31;
  const int h    = lane >> 4;     // half-lane select (K packing / row+8 in C/D)
  const int lrow = lane & 15;
  const int rowBase = (blockIdx.x * WAVES + wave) * 16;

  // ---------------- async stage: global (packed ws) -> LDS, 16B per lane ----------------
  {
    const unsigned char* src = (const unsigned char*)wsH;
    for (int off = tid * 16; off < (int)WSB_STAGE; off += 256 * 16) {
#if __has_builtin(__builtin_amdgcn_global_load_async_to_lds_b128)
      __builtin_amdgcn_global_load_async_to_lds_b128(
          (__attribute__((address_space(1))) v4i*)(src + off),
          (__attribute__((address_space(3))) v4i*)(stage + off),
          0, 0);
#else
      *(float4*)(stage + off) = *(const float4*)(src + off);
#endif
    }
#if __has_builtin(__builtin_amdgcn_s_wait_asynccnt)
    __builtin_amdgcn_s_wait_asynccnt(0);
#else
    asm volatile("s_wait_asynccnt 0" ::: "memory");
#endif
  }

  // ---------------- A-matrix (16x32 f16) from z: 4x b128 loads ----------------
  v16h aZ;
  {
    const float* zr = z + (size_t)(rowBase + lrow) * DIN + h * 8;
    const float4 f0 = *(const float4*)(zr + 0);
    const float4 f1 = *(const float4*)(zr + 4);
    const float4 f2 = *(const float4*)(zr + 16);
    const float4 f3 = *(const float4*)(zr + 20);
    aZ[0]  = (_Float16)f0.x; aZ[1]  = (_Float16)f0.y; aZ[2]  = (_Float16)f0.z; aZ[3]  = (_Float16)f0.w;
    aZ[4]  = (_Float16)f1.x; aZ[5]  = (_Float16)f1.y; aZ[6]  = (_Float16)f1.z; aZ[7]  = (_Float16)f1.w;
    aZ[8]  = (_Float16)f2.x; aZ[9]  = (_Float16)f2.y; aZ[10] = (_Float16)f2.z; aZ[11] = (_Float16)f2.w;
    aZ[12] = (_Float16)f3.x; aZ[13] = (_Float16)f3.y; aZ[14] = (_Float16)f3.z; aZ[15] = (_Float16)f3.w;
  }

  __syncthreads();  // staged LDS visible to all waves

  const _Float16* wst  = (const _Float16*)stage;
  const float*    csqL = (const float*)(stage + WSB_CSQ);

  // ---------------- decoder layer 1: 8 WMMAs -> bias+relu -> LDS ----------------
  {
    const v16h* w1p = (const v16h*)(wst + WH_DECW1);
    #pragma unroll
    for (int n = 0; n < 8; ++n) {
      v16h b = w1p[n * 32 + lane];
      v8f acc = {};
      acc = __builtin_amdgcn_wmma_f32_16x16x32_f16(false, aZ, false, b, (short)0, acc, false, false);
      const float bias = dec_b1[n * 16 + lrow];
      #pragma unroll
      for (int r = 0; r < 8; ++r) {
        float v = fmaxf(acc[r] + bias, 0.0f);
        hbuf[wave][r + h * 8][n * 16 + lrow] = (_Float16)v;  // C/D layout -> row-major LDS
      }
    }
  }
  __syncthreads();

  // ---------------- decoder layer 2: K=128 as 4 WMMAs ----------------
  v8f zf = {};
  {
    const v16h* w2p = (const v16h*)(wst + WH_DECW2);
    #pragma unroll
    for (int kb = 0; kb < 4; ++kb) {
      const v8h lo = *(const v8h*)&hbuf[wave][lrow][kb * 32 + h * 8];       // K = kb*32 + {0..7 | 8..15}
      const v8h hi = *(const v8h*)&hbuf[wave][lrow][kb * 32 + 16 + h * 8];  // K = kb*32 + {16..23 | 24..31}
      v16h a;
      #pragma unroll
      for (int j = 0; j < 8; ++j) { a[j] = lo[j]; a[j + 8] = hi[j]; }
      v16h b = w2p[kb * 32 + lane];
      zf = __builtin_amdgcn_wmma_f32_16x16x32_f16(false, a, false, b, (short)0, zf, false, false);
    }
    const float b2 = dec_b2[lrow];
    #pragma unroll
    for (int r = 0; r < 8; ++r) {
      zf[r] += b2;                                   // zd, kept in f32 for loss/output
      zbuf[wave][r + h * 8][lrow] = (_Float16)zf[r];
    }
  }
  __syncthreads();

  // ---------------- A-matrix for distance GEMM: K=16, zero-padded to 32 ----------------
  v16h aQ;
  {
    const v8h lo = *(const v8h*)&zbuf[wave][lrow][h * 8];
    #pragma unroll
    for (int j = 0; j < 8; ++j) { aQ[j] = lo[j]; aQ[j + 8] = (_Float16)0.0f; }
  }

  // ---------------- distance GEMM over 32 codebook tiles + running argmin ----------------
  float mv[8]; int mi[8];
  #pragma unroll
  for (int r = 0; r < 8; ++r) { mv[r] = 3.0e38f; mi[r] = 0; }
  {
    const v16h* cbp = (const v16h*)(wst + WH_CB);
    #pragma unroll 4
    for (int t = 0; t < NTILES; ++t) {
      v16h b = cbp[t * 32 + lane];
      v8f s = {};
      s = __builtin_amdgcn_wmma_f32_16x16x32_f16(false, aQ, false, b, (short)0, s, false, false);
      const int col = t * 16 + lrow;
      const float cs = csqL[col];
      #pragma unroll
      for (int r = 0; r < 8; ++r) {
        float d = __builtin_fmaf(-2.0f, s[r], cs);   // ||zf||^2 is row-constant: dropped
        if (d < mv[r]) { mv[r] = d; mi[r] = col; }   // ascending t keeps lowest index on ties
      }
    }
  }

  // ---------------- argmin butterfly (16-lane groups), gather z_q, emit ----------------
  float lossAcc = 0.0f;
  #pragma unroll
  for (int r = 0; r < 8; ++r) {
    float v = mv[r]; int id = mi[r];
    { float ov = SWZ_F(v, 1); int oi = SWZ_I(id, 1); if (ov < v || (ov == v && oi < id)) { v = ov; id = oi; } }
    { float ov = SWZ_F(v, 2); int oi = SWZ_I(id, 2); if (ov < v || (ov == v && oi < id)) { v = ov; id = oi; } }
    { float ov = SWZ_F(v, 4); int oi = SWZ_I(id, 4); if (ov < v || (ov == v && oi < id)) { v = ov; id = oi; } }
    { float ov = SWZ_F(v, 8); int oi = SWZ_I(id, 8); if (ov < v || (ov == v && oi < id)) { v = ov; id = oi; } }
    const int row = rowBase + r + h * 8;
    const float zq = codebook[lrow * NE + id];              // z_q[row][lrow]
    out[OUT_ZQ + (size_t)row * LD + lrow] = zq;             // z_q_st == z_q numerically
    const float diff = zq - zf[r];
    lossAcc += diff * diff;
    if (lrow == 0) out[OUT_IDS + row] = (float)id;
  }

  // ---------------- scaler head: 8 WMMAs + fused 128->1 dot ----------------
  {
    float dp[8];
    #pragma unroll
    for (int r = 0; r < 8; ++r) dp[r] = 0.0f;
    const v16h* wp = (const v16h*)(wst + WH_SCW1);
    #pragma unroll
    for (int n = 0; n < 8; ++n) {
      v16h b = wp[n * 32 + lane];
      v8f acc = {};
      acc = __builtin_amdgcn_wmma_f32_16x16x32_f16(false, aZ, false, b, (short)0, acc, false, false);
      const float b1 = sc_b1[n * 16 + lrow];
      const float w2 = sc_w2[n * 16 + lrow];
      #pragma unroll
      for (int r = 0; r < 8; ++r) dp[r] += fmaxf(acc[r] + b1, 0.0f) * w2;
    }
    const float b2 = sc_b2[0];
    #pragma unroll
    for (int r = 0; r < 8; ++r) {
      float v = dp[r];
      v += SWZ_F(v, 1); v += SWZ_F(v, 2); v += SWZ_F(v, 4); v += SWZ_F(v, 8);
      if (lrow == 0) out[OUT_SC + rowBase + r + h * 8] = v + b2;
    }
  }

  // ---------------- redshift head: same, with output ReLU ----------------
  {
    float dp[8];
    #pragma unroll
    for (int r = 0; r < 8; ++r) dp[r] = 0.0f;
    const v16h* wp = (const v16h*)(wst + WH_RSW1);
    #pragma unroll
    for (int n = 0; n < 8; ++n) {
      v16h b = wp[n * 32 + lane];
      v8f acc = {};
      acc = __builtin_amdgcn_wmma_f32_16x16x32_f16(false, aZ, false, b, (short)0, acc, false, false);
      const float b1 = rs_b1[n * 16 + lrow];
      const float w2 = rs_w2[n * 16 + lrow];
      #pragma unroll
      for (int r = 0; r < 8; ++r) dp[r] += fmaxf(acc[r] + b1, 0.0f) * w2;
    }
    const float b2 = rs_b2[0];
    #pragma unroll
    for (int r = 0; r < 8; ++r) {
      float v = dp[r];
      v += SWZ_F(v, 1); v += SWZ_F(v, 2); v += SWZ_F(v, 4); v += SWZ_F(v, 8);
      if (lrow == 0) out[OUT_RS + rowBase + r + h * 8] = fmaxf(v + b2, 0.0f);
    }
  }

  // ---------------- deterministic block-level loss partial ----------------
  lossAcc += SWZ_F(lossAcc, 1);
  lossAcc += SWZ_F(lossAcc, 2);
  lossAcc += SWZ_F(lossAcc, 4);
  lossAcc += SWZ_F(lossAcc, 8);
  lossAcc += SWZ_F(lossAcc, 16);
  if (lane == 0) wsum[wave] = lossAcc;
  __syncthreads();
  if (tid == 0) {
    float s = 0.0f;
    #pragma unroll
    for (int w = 0; w < WAVES; ++w) s += wsum[w];
    partials[blockIdx.x] = s;
  }
}

// ---------------------------------------------------------------------------
// Final reduction: codebook_loss = (1+BETA) * mean((z_q - zd)^2)
// ---------------------------------------------------------------------------
__global__ void loss_kernel(const float* __restrict__ partials, int n, float* __restrict__ out) {
  __shared__ float s[256];
  float acc = 0.0f;
  for (int i = threadIdx.x; i < n; i += 256) acc += partials[i];
  s[threadIdx.x] = acc;
  __syncthreads();
  for (int st = 128; st > 0; st >>= 1) {
    if (threadIdx.x < st) s[threadIdx.x] += s[threadIdx.x + st];
    __syncthreads();
  }
  if (threadIdx.x == 0) out[OUT_LOSS] = s[0] * (1.25f / ((float)BROWS * (float)LD));
}

// ---------------------------------------------------------------------------
extern "C" void kernel_launch(void* const* d_in, const int* in_sizes, int n_in,
                              void* d_out, int out_size, void* d_ws, size_t ws_size,
                              hipStream_t stream) {
  const float* z        = (const float*)d_in[0];
  const float* codebook = (const float*)d_in[1];
  const float* dec_w1   = (const float*)d_in[2];
  const float* dec_b1   = (const float*)d_in[3];
  const float* dec_w2   = (const float*)d_in[4];
  const float* dec_b2   = (const float*)d_in[5];
  const float* sc_w1    = (const float*)d_in[6];
  const float* sc_b1    = (const float*)d_in[7];
  const float* sc_w2    = (const float*)d_in[8];
  const float* sc_b2    = (const float*)d_in[9];
  const float* rs_w1    = (const float*)d_in[10];
  const float* rs_b1    = (const float*)d_in[11];
  const float* rs_w2    = (const float*)d_in[12];
  const float* rs_b2    = (const float*)d_in[13];

  float*     out  = (float*)d_out;
  _Float16*  wsH  = (_Float16*)d_ws;
  float*     csq  = (float*)((char*)d_ws + WSB_CSQ);
  float*     part = (float*)((char*)d_ws + WSB_PART);

  prep_kernel<<<64, 256, 0, stream>>>(dec_w1, sc_w1, rs_w1, dec_w2, codebook, wsH, csq);
  vq_kernel<<<NBLOCKS, 256, 0, stream>>>(z, codebook, dec_b1, dec_b2,
                                         sc_b1, sc_w2, sc_b2,
                                         rs_b1, rs_w2, rs_b2,
                                         wsH, out, part);
  loss_kernel<<<1, 256, 0, stream>>>(part, NBLOCKS, out);
}
